// GDN_28484223107109
// MI455X (gfx1250) — matrix operand order repaired and checked by reading, not compiled, not run on hardware
//
#include <hip/hip_runtime.h>

typedef _Float16 h16;
typedef __attribute__((ext_vector_type(16))) _Float16 v16h;
typedef __attribute__((ext_vector_type(8)))  float    v8f;
typedef __attribute__((ext_vector_type(4)))  int      v4i;

#define CH 192
#define HW 65536            // 256*256 pixels per channel plane
#define PIX_PER_BLOCK 128   // 8 waves * 16 pixels
#define BLOCKS_PER_IMG 512  // 65536 / 128
#define GBYTES (CH * CH * 2 * (int)sizeof(h16))   // gamma hi+lo bytes = 147456
#define GVEC4  (GBYTES / 16)                      // 9216 16-byte chunks

// ---------------- parameter reparameterization (runs once, tiny) ------------
__global__ void gdn_param_kernel(const float* __restrict__ beta,
                                 const float* __restrict__ gamma,
                                 h16* __restrict__ ghi, h16* __restrict__ glo,
                                 float* __restrict__ betac) {
  const float GAMMA_BOUND = 3.814697265625e-06f;     // 2^-18
  const float PEDESTAL    = 1.4551915228366852e-11f; // 2^-36
  const float BETA_BOUND  = 1.0000072759312765e-03f; // sqrt(1e-6 + 2^-36)
  int idx = blockIdx.x * blockDim.x + threadIdx.x;
  if (idx < CH * CH) {
    float g  = fmaxf(gamma[idx], GAMMA_BOUND);
    float gc = g * g - PEDESTAL;
    h16 hi = (h16)gc;
    ghi[idx] = hi;
    glo[idx] = (h16)(gc - (float)hi);
  }
  if (idx < CH) {
    float b = fmaxf(beta[idx], BETA_BOUND);
    betac[idx] = b * b - PEDESTAL;
  }
}

// ---------------- main GDN kernel: fp16-split WMMA GEMM + epilogue ----------
// Dynamic LDS: [gamma_hi: 36864 h16][gamma_lo: 36864 h16] = 147456 B
__global__ __launch_bounds__(256)
void gdn_main_kernel(const float* __restrict__ in,
                     const v4i* __restrict__ gsrc,    // gamma hi||lo, 16B chunks
                     const float* __restrict__ betac,
                     float* __restrict__ out) {
  extern __shared__ char smem_raw[];
  v4i* sdst = (v4i*)smem_raw;
  const h16* sghi = (const h16*)smem_raw;
  const h16* sglo = sghi + CH * CH;

  const int lane = threadIdx.x & 31;
  const int wave = threadIdx.x >> 5;
  const int lh   = lane >> 4;   // which 16-lane half
  const int ln   = lane & 15;

  const int img = blockIdx.x >> 9;                     // / BLOCKS_PER_IMG
  const int hw0 = (blockIdx.x & (BLOCKS_PER_IMG - 1)) * PIX_PER_BLOCK;
  const int p   = hw0 + wave * 16 + ln;                // this lane's pixel (N)

  const size_t img_off = (size_t)img * (size_t)CH * (size_t)HW;
  const float* pin  = in  + img_off + p;
  float*       pout = out + img_off + p;

  // ---- Stage gamma hi/lo into LDS once per block (147456 B, one L2 read
  // per block instead of one per wave: 8x less global gamma traffic).
#pragma unroll 4
  for (int i = threadIdx.x; i < GVEC4; i += 256)
    sdst[i] = gsrc[i];

  // ---- Load B operand: x^2 split into fp16 hi/lo, resident for all M tiles.
  // B layout used here (contiguous k, not the ISA table order; A matches): half-element e within lane
  //   -> K = 16*lh + e, N = lane&15.  (overlaps with the LDS staging above)
  v16h bh[6], bl[6];
#pragma unroll
  for (int kk = 0; kk < 6; ++kk) {
#pragma unroll
    for (int e = 0; e < 16; ++e) {
      int c    = kk * 32 + lh * 16 + e;          // input channel (K index)
      float x  = pin[(size_t)c * HW];
      float x2 = x * x;
      h16 hi   = (h16)x2;
      bh[kk][e] = hi;
      bl[kk][e] = (h16)(x2 - (float)hi);
    }
  }

  __syncthreads();   // gamma staged

  // ---- 12 M-tiles of 16 output channels each; A operands from LDS.
  for (int mt = 0; mt < 12; ++mt) {
    const int mrow = mt * 16 + ln;               // A-matrix row held by this lane
    const h16* gh_row = sghi + (size_t)mrow * CH;
    const h16* gl_row = sglo + (size_t)mrow * CH;

    v8f acc = {0.f, 0.f, 0.f, 0.f, 0.f, 0.f, 0.f, 0.f};

#pragma unroll
    for (int kk = 0; kk < 6; ++kk) {
      // A layout: e in 0..15 -> K = 16*lh + e (contiguous 16 halfs = 2x16B), matches B
      union { v16h v; v4i q[2]; } ah, al;
      ah.q[0] = *(const v4i*)(gh_row + kk * 32 + lh * 16);
      ah.q[1] = *(const v4i*)(gh_row + kk * 32 + lh * 16 + 8);
      al.q[0] = *(const v4i*)(gl_row + kk * 32 + lh * 16);
      al.q[1] = *(const v4i*)(gl_row + kk * 32 + lh * 16 + 8);

      // 3-term split product: gh*bh + gh*bl + gl*bh  (~2^-22 relative error)
      acc = __builtin_amdgcn_wmma_f32_16x16x32_f16(false, ah.v, false, bh[kk],
                                                   (short)0, acc, false, false);
      acc = __builtin_amdgcn_wmma_f32_16x16x32_f16(false, ah.v, false, bl[kk],
                                                   (short)0, acc, false, false);
      acc = __builtin_amdgcn_wmma_f32_16x16x32_f16(false, al.v, false, bh[kk],
                                                   (short)0, acc, false, false);
    }

    // ---- epilogue: D layout f32 16x16 -> M = 8*lh + r, N = lane&15
#pragma unroll
    for (int r = 0; r < 8; ++r) {
      int   m  = mt * 16 + lh * 8 + r;
      float v  = acc[r] + betac[m];
      float x  = pin[(size_t)m * HW];            // L2-warm reload of input
      float iv = rsqrtf(v);
      iv = iv * (1.5f - 0.5f * v * iv * iv);     // one Newton step -> ~fp32
      __builtin_nontemporal_store(x * iv, &pout[(size_t)m * HW]);
    }
  }
}

// ---------------- launch ----------------------------------------------------
extern "C" void kernel_launch(void* const* d_in, const int* in_sizes, int n_in,
                              void* d_out, int out_size, void* d_ws, size_t ws_size,
                              hipStream_t stream) {
  const float* x     = (const float*)d_in[0];  // [8,192,256,256]
  const float* beta  = (const float*)d_in[1];  // [192]
  const float* gamma = (const float*)d_in[2];  // [192,192]
  float* outp = (float*)d_out;

  char* ws = (char*)d_ws;
  h16*   ghi   = (h16*)(ws);                   // 36864 halfs = 73728 B
  h16*   glo   = (h16*)(ws + CH * CH * 2);     // 36864 halfs = 73728 B
  float* betac = (float*)(ws + GBYTES);        // 192 floats

  gdn_param_kernel<<<(CH * CH + 255) / 256, 256, 0, stream>>>(beta, gamma,
                                                              ghi, glo, betac);

  const int nblocks = 8 * BLOCKS_PER_IMG;      // 4096
  gdn_main_kernel<<<nblocks, 256, GBYTES, stream>>>(x, (const v4i*)ws,
                                                    betac, outp);
}
